// DynamicSeq2Seq_2748779069735
// MI455X (gfx1250) — compile-verified, matrix operand order checked by reading, not run on hardware
//
#include <hip/hip_runtime.h>
#include <hip/hip_bf16.h>
#include <math.h>
#include <stdint.h>

typedef __attribute__((ext_vector_type(16))) __bf16 v16bf;
typedef __attribute__((ext_vector_type(8)))  __bf16 v8bf;
typedef __attribute__((ext_vector_type(8)))  float  v8f;

// problem constants
#define Bq 128
#define Sq 64
#define TD 10
#define Eq 256
#define Hq 512
#define Lq 512
#define Vq 32000
#define G4H 2048
#define START_TOK 2

// ---------------- elementwise / setup kernels ----------------

__global__ void k_cast_bf16(const float* __restrict__ src, __bf16* __restrict__ dst, int n) {
  int i = blockIdx.x * blockDim.x + threadIdx.x;
  if (i < n) dst[i] = (__bf16)src[i];
}

// dst[rows, k1+k2] = [w1 | w2] cast to bf16
__global__ void k_concat2_bf16(const float* __restrict__ w1, int k1,
                               const float* __restrict__ w2, int k2,
                               __bf16* __restrict__ dst, int rows) {
  int i = blockIdx.x * blockDim.x + threadIdx.x;
  int kt = k1 + k2;
  long n = (long)rows * kt;
  if (i >= n) return;
  int r = i / kt, c = i - r * kt;
  float v = (c < k1) ? w1[(long)r * k1 + c] : w2[(long)r * k2 + (c - k1)];
  dst[i] = (__bf16)v;
}

__global__ void k_zero_f32(float* __restrict__ p, int n) {
  int i = blockIdx.x * blockDim.x + threadIdx.x;
  if (i < n) p[i] = 0.0f;
}

__global__ void k_zero_bf16(__bf16* __restrict__ p, int n) {
  int i = blockIdx.x * blockDim.x + threadIdx.x;
  if (i < n) p[i] = (__bf16)0.0f;
}

// encoder embedding gather for timestep t -> A_enc[:, 0:E]
__global__ void k_enc_slice(const int* __restrict__ x, const float* __restrict__ emb,
                            __bf16* __restrict__ A, int t) {
  int i = blockIdx.x * blockDim.x + threadIdx.x; // B*E
  if (i >= Bq * Eq) return;
  int b = i >> 8;         // /E (E=256)
  int e = i & (Eq - 1);
  int tok = x[b * Sq + t];
  A[(long)b * (Eq + Hq) + e] = (__bf16)emb[(long)tok * Eq + e];
}

// decoder embedding gather (teacher forcing) for step t -> A_dec[:, 0:E]
__global__ void k_dec_slice(const int* __restrict__ y, const float* __restrict__ emb,
                            __bf16* __restrict__ A, int t) {
  int i = blockIdx.x * blockDim.x + threadIdx.x; // B*E
  if (i >= Bq * Eq) return;
  int b = i >> 8;
  int e = i & (Eq - 1);
  int tok = (t == 0) ? START_TOK : y[b * TD + (t - 1)];
  A[(long)b * (Eq + 2 * Hq) + e] = (__bf16)emb[(long)tok * Eq + e];
}

__device__ __forceinline__ float sigf(float x) { return 1.0f / (1.0f + __expf(-x)); }

// LSTM pointwise: gate order i,f,g,o. Updates c (f32 state), writes h to bf16 buffers.
__global__ void k_lstm_pw(const float* __restrict__ gates,
                          const float* __restrict__ bih, const float* __restrict__ bhh,
                          float* __restrict__ c, __bf16* __restrict__ h_bf,
                          __bf16* __restrict__ hdst2, int stride2,   // A_enc h slice (or null)
                          float* __restrict__ ho_f32, long stride3,  // enc_output f32 slice (or null)
                          __bf16* __restrict__ ho_bf, long stride4)  // enc_output bf16 slice (or null)
{
  int i = blockIdx.x * blockDim.x + threadIdx.x; // B*H
  if (i >= Bq * Hq) return;
  int b = i >> 9;          // /H (H=512)
  int j = i & (Hq - 1);
  const float* g = gates + (long)b * G4H;
  float gi = g[j]            + bih[j]            + bhh[j];
  float gf = g[Hq + j]       + bih[Hq + j]       + bhh[Hq + j];
  float gg = g[2 * Hq + j]   + bih[2 * Hq + j]   + bhh[2 * Hq + j];
  float go = g[3 * Hq + j]   + bih[3 * Hq + j]   + bhh[3 * Hq + j];
  float cv = sigf(gf) * c[i] + sigf(gi) * tanhf(gg);
  float hv = sigf(go) * tanhf(cv);
  c[i] = cv;
  h_bf[i] = (__bf16)hv;
  if (hdst2) hdst2[(long)b * stride2 + j] = (__bf16)hv;
  if (ho_f32) ho_f32[(long)b * stride3 + j] = hv;
  if (ho_bf)  ho_bf[(long)b * stride4 + j] = (__bf16)hv;
}

// scores[b,s] = sum_h tanh(Uhj[b,h] + Ws[b,s,h]) * v[h]   (one wave per (b,s))
__global__ void k_attn_score(const float* __restrict__ Uhj, const float* __restrict__ Ws,
                             const float* __restrict__ vvec, float* __restrict__ scores) {
  int gt = blockIdx.x * blockDim.x + threadIdx.x;
  int w = gt >> 5;
  int lane = gt & 31;
  if (w >= Bq * Sq) return;
  int b = w >> 6;          // /S (S=64)
  int s = w & (Sq - 1);
  const float* u   = Uhj + (long)b * Hq;
  const float* wsr = Ws  + ((long)b * Sq + s) * Hq;
  float acc = 0.0f;
#pragma unroll 4
  for (int h = lane; h < Hq; h += 32)
    acc += tanhf(u[h] + wsr[h]) * vvec[h];
#pragma unroll
  for (int m = 16; m >= 1; m >>= 1) acc += __shfl_xor(acc, m, 32);
  if (lane == 0) scores[w] = acc;
}

// softmax over S=64 per batch row; also writes attn_w output row t. One wave per b.
__global__ void k_softmax(const float* __restrict__ scores, float* __restrict__ attn,
                          float* __restrict__ out_attn, int t) {
  int gt = blockIdx.x * blockDim.x + threadIdx.x;
  int b = gt >> 5;
  int lane = gt & 31;
  if (b >= Bq) return;
  float s0 = scores[b * Sq + lane];
  float s1 = scores[b * Sq + 32 + lane];
  float mx = fmaxf(s0, s1);
#pragma unroll
  for (int m = 16; m >= 1; m >>= 1) mx = fmaxf(mx, __shfl_xor(mx, m, 32));
  float e0 = __expf(s0 - mx), e1 = __expf(s1 - mx);
  float sum = e0 + e1;
#pragma unroll
  for (int m = 16; m >= 1; m >>= 1) sum += __shfl_xor(sum, m, 32);
  float inv = 1.0f / sum;
  attn[b * Sq + lane]      = e0 * inv;
  attn[b * Sq + 32 + lane] = e1 * inv;
  long ob = (long)b * Sq * Sq + (long)t * Sq;
  out_attn[ob + lane]      = e0 * inv;
  out_attn[ob + 32 + lane] = e1 * inv;
}

// context[b,h] = sum_s attn[b,s]*enc_out[b,s,h]; written (bf16) into BOTH A_dec segments
__global__ void k_context(const float* __restrict__ attn, const float* __restrict__ enc_out,
                          __bf16* __restrict__ A) {
  int i = blockIdx.x * blockDim.x + threadIdx.x; // B*H
  if (i >= Bq * Hq) return;
  int b = i >> 9;
  int h = i & (Hq - 1);
  const float* a  = attn + b * Sq;
  const float* eo = enc_out + (long)b * Sq * Hq + h;
  float acc = 0.0f;
#pragma unroll 4
  for (int s = 0; s < Sq; ++s) acc += a[s] * eo[(long)s * Hq];
  __bf16 cv = (__bf16)acc;
  long rb = (long)b * (Eq + 2 * Hq);
  A[rb + Eq + h]      = cv;
  A[rb + Eq + Hq + h] = cv;
}

// ---------------- WMMA GEMM with async global->LDS double buffering ----------------
// C[M,N] = act(A[M,K] * W[N,K]^T + bias), A/W bf16 row-major (both K-contiguous).
// M%128==0, N%64==0, K%64==0.

#define BM 128
#define BN 64
#define BK 64   // two 32-wide WMMA K-slices per pipeline stage

// async global -> LDS, 16 bytes per lane (ASYNCcnt-tracked, VGPR-file bypass)
__device__ __forceinline__ void async_b128(uint32_t lds_off, const __bf16* gptr) {
  asm volatile("global_load_async_to_lds_b128 %0, %1, off"
               :: "v"(lds_off), "v"(gptr) : "memory");
}
#define S_WAIT_ASYNC(n) asm volatile("s_wait_asynccnt " #n ::: "memory")

// Load a 16x32 bf16 fragment from a row-major LDS tile (pitch BK).
// ISA 16-bit A/B layout: row/col = lane&15; lanes<16 take K 0..7,16..23;
// lanes>=16 take K 8..15,24..31 (kb=(lane>>4)*8; halves at kb and kb+16).
__device__ __forceinline__ v16bf frag_ld(const __bf16* base, int lane) {
  int r  = lane & 15;
  int kb = (lane >> 4) << 3;
  const __bf16* p = base + r * BK + kb;
  v8bf lo = *(const v8bf*)p;
  v8bf hi = *(const v8bf*)(p + 16);
  v16bf f;
#pragma unroll
  for (int i = 0; i < 8; ++i) { f[i] = lo[i]; f[i + 8] = hi[i]; }
  return f;
}

__global__ __launch_bounds__(256)
void k_gemm_bf16(const __bf16* __restrict__ A, const __bf16* __restrict__ W,
                 int M, int N, int K,
                 float* __restrict__ C, long ldc,
                 __bf16* __restrict__ Cbf, long ldcb,
                 const float* __restrict__ bias, int act) {
  __shared__ __bf16 sA[2][BM * BK];   // 2 x 16 KB
  __shared__ __bf16 sB[2][BN * BK];   // 2 x 8 KB
  const int tid  = threadIdx.x;
  const int lane = tid & 31;
  const int wave = tid >> 5;
  const int bm = blockIdx.y * BM;
  const int bn = blockIdx.x * BN;
  const int wm = (wave & 3) * 32;   // 4 waves along M
  const int wn = (wave >> 2) * 32;  // 2 waves along N

  // per-thread async-copy assignments (16B vectors, BK/8 = 8 vectors per row)
  // A tile: 1024 vectors -> 4/thread; B tile: 512 vectors -> 2/thread
  int aRow[4], aCol[4], bRow[2], bCol[2];
  uint32_t aOff[4], bOff[2];
  const uint32_t aBase = (uint32_t)(uintptr_t)&sA[0][0];
  const uint32_t bBase = (uint32_t)(uintptr_t)&sB[0][0];
  const uint32_t aStride = BM * BK * 2;  // bytes per buffer
  const uint32_t bStride = BN * BK * 2;
#pragma unroll
  for (int v = 0; v < 4; ++v) {
    int id = tid * 4 + v;
    aRow[v] = id >> 3;
    aCol[v] = (id & 7) << 3;
    aOff[v] = aBase + (uint32_t)(aRow[v] * BK + aCol[v]) * 2u;
  }
#pragma unroll
  for (int v = 0; v < 2; ++v) {
    int id = tid * 2 + v;
    bRow[v] = id >> 3;
    bCol[v] = (id & 7) << 3;
    bOff[v] = bBase + (uint32_t)(bRow[v] * BK + bCol[v]) * 2u;
  }

  auto issue = [&](int k0, int bb) {
#pragma unroll
    for (int v = 0; v < 4; ++v)
      async_b128(aOff[v] + (uint32_t)bb * aStride, A + (long)(bm + aRow[v]) * K + k0 + aCol[v]);
#pragma unroll
    for (int v = 0; v < 2; ++v)
      async_b128(bOff[v] + (uint32_t)bb * bStride, W + (long)(bn + bRow[v]) * K + k0 + bCol[v]);
  };

  const int nk = K / BK;
  issue(0, 0);
  if (nk > 1) issue(BK, 1);

  v8f acc[2][2] = {};

  for (int kt = 0; kt < nk; ++kt) {
    const int cur = kt & 1;
    // wait for this stage's 6 async ops (in-order completion); next stage may remain in flight
    if (kt + 1 < nk) { S_WAIT_ASYNC(0x6); } else { S_WAIT_ASYNC(0x0); }
    __syncthreads();   // all waves' stage-kt data landed in LDS

    const __bf16* pa = sA[cur];
    const __bf16* pb = sB[cur];
#pragma unroll
    for (int ks = 0; ks < 2; ++ks) {
      const int ko = ks * 32;
      v16bf a0 = frag_ld(pa + (wm)      * BK + ko, lane);
      v16bf a1 = frag_ld(pa + (wm + 16) * BK + ko, lane);
      v16bf b0 = frag_ld(pb + (wn)      * BK + ko, lane);
      v16bf b1 = frag_ld(pb + (wn + 16) * BK + ko, lane);
      acc[0][0] = __builtin_amdgcn_wmma_f32_16x16x32_bf16(false, a0, false, b0, (short)0, acc[0][0], false, false);
      acc[0][1] = __builtin_amdgcn_wmma_f32_16x16x32_bf16(false, a0, false, b1, (short)0, acc[0][1], false, false);
      acc[1][0] = __builtin_amdgcn_wmma_f32_16x16x32_bf16(false, a1, false, b0, (short)0, acc[1][0], false, false);
      acc[1][1] = __builtin_amdgcn_wmma_f32_16x16x32_bf16(false, a1, false, b1, (short)0, acc[1][1], false, false);
    }

    __syncthreads();   // all waves done reading buffer `cur` before it is refilled
    if (kt + 2 < nk) issue((kt + 2) * BK, cur);
  }

  // epilogue: C/D layout — VGPR i: M = (lane>=16 ? 8 : 0) + i, N = lane&15
  const int n_in = lane & 15;
  const int m8   = (lane >> 4) << 3;
#pragma unroll
  for (int mi = 0; mi < 2; ++mi) {
#pragma unroll
    for (int ni = 0; ni < 2; ++ni) {
      int gn  = bn + wn + ni * 16 + n_in;
      int gm0 = bm + wm + mi * 16 + m8;
      float bv = bias ? bias[gn] : 0.0f;
#pragma unroll
      for (int i = 0; i < 8; ++i) {
        float v = acc[mi][ni][i] + bv;
        if (act == 1) v = tanhf(v);
        if (C)   C[(long)(gm0 + i) * ldc + gn] = v;
        if (Cbf) Cbf[(long)(gm0 + i) * ldcb + gn] = (__bf16)v;
      }
    }
  }
}

static inline void gemm(hipStream_t s, const __bf16* A, const __bf16* W,
                        int M, int N, int K,
                        float* C, long ldc, __bf16* Cbf, long ldcb,
                        const float* bias, int act) {
  dim3 g(N / BN, M / BM);
  k_gemm_bf16<<<g, 256, 0, s>>>(A, W, M, N, K, C, ldc, Cbf, ldcb, bias, act);
}

// ---------------- host orchestration ----------------

extern "C" void kernel_launch(void* const* d_in, const int* in_sizes, int n_in,
                              void* d_out, int out_size, void* d_ws, size_t ws_size,
                              hipStream_t stream) {
  const int*   x          = (const int*)d_in[0];
  const int*   y          = (const int*)d_in[1];
  const float* enc_emb    = (const float*)d_in[2];
  const float* enc_Wih    = (const float*)d_in[3];
  const float* enc_Whh    = (const float*)d_in[4];
  const float* enc_bih    = (const float*)d_in[5];
  const float* enc_bhh    = (const float*)d_in[6];
  const float* dec_emb    = (const float*)d_in[7];
  const float* dec_Wih    = (const float*)d_in[8];
  const float* dec_Whh    = (const float*)d_in[9];
  const float* dec_bih    = (const float*)d_in[10];
  const float* dec_bhh    = (const float*)d_in[11];
  const float* attn_enc_w = (const float*)d_in[12];
  const float* attn_enc_b = (const float*)d_in[13];
  const float* attn_dec_w = (const float*)d_in[14];
  const float* attn_dec_b = (const float*)d_in[15];
  const float* attn_v     = (const float*)d_in[16];
  const float* fc1_w      = (const float*)d_in[17];
  const float* fc1_b      = (const float*)d_in[18];
  const float* fc2_w      = (const float*)d_in[19];
  const float* fc2_b      = (const float*)d_in[20];

  float* out_logits = (float*)d_out;                         // [B, TD, V]
  float* out_attn   = out_logits + (size_t)Bq * TD * Vq;     // [B, S, S]

  // workspace carve-up (256B aligned)
  char* p = (char*)d_ws;
  auto alloc = [&](size_t bytes) { char* r = p; p += (bytes + 255) & ~(size_t)255; return r; };
  __bf16* encW    = (__bf16*)alloc((size_t)G4H * (Eq + Hq) * 2);       // [4H, E+H]  = [Wih|Whh]
  __bf16* decW    = (__bf16*)alloc((size_t)G4H * (Eq + 2 * Hq) * 2);   // [4H, E+2H] = [Wih|Whh]
  __bf16* aew     = (__bf16*)alloc((size_t)Hq * Hq * 2);
  __bf16* adw     = (__bf16*)alloc((size_t)Hq * Hq * 2);
  __bf16* f1w     = (__bf16*)alloc((size_t)Lq * Hq * 2);
  __bf16* f2w     = (__bf16*)alloc((size_t)Vq * Lq * 2);
  __bf16* Aenc    = (__bf16*)alloc((size_t)Bq * (Eq + Hq) * 2);        // [B, E+H]
  __bf16* Adec    = (__bf16*)alloc((size_t)Bq * (Eq + 2 * Hq) * 2);    // [B, E+2H]
  __bf16* hbf     = (__bf16*)alloc((size_t)Bq * Hq * 2);
  float*  cbuf    = (float*)alloc((size_t)Bq * Hq * 4);
  float*  gates   = (float*)alloc((size_t)Bq * G4H * 4);
  float*  encOut  = (float*)alloc((size_t)Bq * Sq * Hq * 4);
  __bf16* encOutB = (__bf16*)alloc((size_t)Bq * Sq * Hq * 2);
  float*  WsBuf   = (float*)alloc((size_t)Bq * Sq * Hq * 4);
  float*  Uhj     = (float*)alloc((size_t)Bq * Hq * 4);
  float*  scores  = (float*)alloc((size_t)Bq * Sq * 4);
  float*  attn    = (float*)alloc((size_t)Bq * Sq * 4);
  __bf16* t1bf    = (__bf16*)alloc((size_t)Bq * Lq * 2);

  auto cdiv = [](long a, long b) { return (int)((a + b - 1) / b); };

  // weight casts (once per call)
  k_concat2_bf16<<<cdiv((long)G4H * (Eq + Hq), 256), 256, 0, stream>>>(enc_Wih, Eq, enc_Whh, Hq, encW, G4H);
  k_concat2_bf16<<<cdiv((long)G4H * (Eq + 2 * Hq), 256), 256, 0, stream>>>(dec_Wih, Eq + Hq, dec_Whh, Hq, decW, G4H);
  k_cast_bf16<<<cdiv(Hq * Hq, 256), 256, 0, stream>>>(attn_enc_w, aew, Hq * Hq);
  k_cast_bf16<<<cdiv(Hq * Hq, 256), 256, 0, stream>>>(attn_dec_w, adw, Hq * Hq);
  k_cast_bf16<<<cdiv(Lq * Hq, 256), 256, 0, stream>>>(fc1_w, f1w, Lq * Hq);
  k_cast_bf16<<<cdiv((long)Vq * Lq, 256), 256, 0, stream>>>(fc2_w, f2w, Vq * Lq);

  // state init: h=0, c=0, zero attn output plane
  k_zero_bf16<<<cdiv(Bq * (Eq + Hq), 256), 256, 0, stream>>>(Aenc, Bq * (Eq + Hq));
  k_zero_bf16<<<cdiv(Bq * Hq, 256), 256, 0, stream>>>(hbf, Bq * Hq);
  k_zero_f32<<<cdiv(Bq * Hq, 256), 256, 0, stream>>>(cbuf, Bq * Hq);
  k_zero_f32<<<cdiv((long)Bq * Sq * Sq, 256), 256, 0, stream>>>(out_attn, Bq * Sq * Sq);

  // ---- encoder: 64 recurrent steps, fused [xe|h] x [Wih|Whh]^T ----
  for (int t = 0; t < Sq; ++t) {
    k_enc_slice<<<cdiv(Bq * Eq, 256), 256, 0, stream>>>(x, enc_emb, Aenc, t);
    gemm(stream, Aenc, encW, Bq, G4H, Eq + Hq, gates, G4H, nullptr, 0, nullptr, 0);
    k_lstm_pw<<<cdiv(Bq * Hq, 256), 256, 0, stream>>>(
        gates, enc_bih, enc_bhh, cbuf, hbf,
        Aenc + Eq, Eq + Hq,
        encOut + (long)t * Hq, (long)Sq * Hq,
        encOutB + (long)t * Hq, (long)Sq * Hq);
  }

  // ---- hoisted attention keys: Ws = enc_out @ attn_enc_w^T + b  (M=8192) ----
  gemm(stream, encOutB, aew, Bq * Sq, Hq, Hq, WsBuf, Hq, nullptr, 0, attn_enc_b, 0);

  // ---- decoder: 10 teacher-forced steps ----
  for (int t = 0; t < TD; ++t) {
    gemm(stream, hbf, adw, Bq, Hq, Hq, Uhj, Hq, nullptr, 0, attn_dec_b, 0);
    k_dec_slice<<<cdiv(Bq * Eq, 256), 256, 0, stream>>>(y, dec_emb, Adec, t);
    k_attn_score<<<cdiv((long)Bq * Sq * 32, 256), 256, 0, stream>>>(Uhj, WsBuf, attn_v, scores);
    k_softmax<<<cdiv(Bq * 32, 256), 256, 0, stream>>>(scores, attn, out_attn, t);
    k_context<<<cdiv(Bq * Hq, 256), 256, 0, stream>>>(attn, encOut, Adec);
    // gates = [emb|ctx|ctx] x [dec_Wih|dec_Whh]^T   (h was overwritten by context)
    gemm(stream, Adec, decW, Bq, G4H, Eq + 2 * Hq, gates, G4H, nullptr, 0, nullptr, 0);
    k_lstm_pw<<<cdiv(Bq * Hq, 256), 256, 0, stream>>>(
        gates, dec_bih, dec_bhh, cbuf, hbf,
        (__bf16*)nullptr, 0, (float*)nullptr, 0, (__bf16*)nullptr, 0);
    // t1 = tanh(h @ fc1^T + b)  (bf16 out), then logits = t1 @ fc2^T + b -> d_out slice
    gemm(stream, hbf, f1w, Bq, Lq, Hq, nullptr, 0, t1bf, Lq, fc1_b, 1);
    gemm(stream, t1bf, f2w, Bq, Vq, Lq, out_logits + (long)t * Vq, (long)TD * Vq,
         nullptr, 0, fc2_b, 0);
  }
}
